// PLEModel_4604204941530
// MI455X (gfx1250) — compile-verified, matrix operand order checked by reading, not run on hardware
//
#include <hip/hip_runtime.h>

// ---------------------------------------------------------------------------
// PLE / CGC multi-task model on MI455X (gfx1250, wave32, WMMA).
// Expert GEMMs: v_wmma_f32_16x16x32_bf16, fp32 accumulate.
// Weights are converted fp32 -> bf16 AND transposed to (E, N, K) so both A and
// B fragments load as contiguous 16-byte global_load_b128 chunks.
// ---------------------------------------------------------------------------

#define PLE_B 16384

typedef __bf16 bf16_t;
typedef bf16_t       v16bf __attribute__((ext_vector_type(16)));
typedef float        v8f   __attribute__((ext_vector_type(8)));
typedef unsigned int v4u   __attribute__((ext_vector_type(4)));

union BF16Frag {
  v16bf          v;
  unsigned short u[16];
  v4u            q[2];
};

__device__ __forceinline__ unsigned short f2bf(float f) {
  unsigned u = __float_as_uint(f);
  u += 0x7FFFu + ((u >> 16) & 1u);   // round-to-nearest-even
  return (unsigned short)(u >> 16);
}

__device__ __forceinline__ float bf2f(unsigned short h) {
  return __uint_as_float(((unsigned)h) << 16);
}

// ---------------------------------------------------------------------------
// fp32 -> bf16 (plain)
// ---------------------------------------------------------------------------
__global__ __launch_bounds__(256) void cvt_f32_to_bf16(
    const float* __restrict__ src, unsigned short* __restrict__ dst, int n) {
  int i = blockIdx.x * blockDim.x + threadIdx.x;
  if (i < n) dst[i] = f2bf(src[i]);
}

// ---------------------------------------------------------------------------
// fp32 (E,K,N) -> bf16 (E,N,K) transpose. Coalesced writes along K.
// ---------------------------------------------------------------------------
__global__ __launch_bounds__(256) void cvt_transpose_f32_bf16(
    const float* __restrict__ src, unsigned short* __restrict__ dst,
    int K, int N, int total /* = E*K*N */) {
  int idx = blockIdx.x * blockDim.x + threadIdx.x;
  if (idx >= total) return;
  const int nk = N * K;
  const int e  = idx / nk;
  const int r  = idx - e * nk;
  const int n  = r / K;
  const int k  = r - n * K;
  dst[idx] = f2bf(src[(size_t)e * nk + (size_t)k * N + n]);
}

// ---------------------------------------------------------------------------
// bf16 GEMM + bias + ReLU via WMMA.
//   Y[b, e*N + n] = relu( sum_k X[b,k] * W[e,k,n] + bias[e,n] )
// Wt is the TRANSPOSED weight: (E, N, K) bf16 row-major.
// Grid: (B/128, N/64, E). Block: 128 threads = 4 waves.
// Each wave computes a 32(M) x 64(N) strip: 2 A frags x 4 B frags
// -> 8 v_wmma_f32_16x16x32_bf16 per K=32 step; every fragment is two
// contiguous 16-byte loads (ISA 7.12.2 16-bit A/B layout: lane = M|N mod 16,
// K-half = lane>>4, 8 consecutive K per half).
// ---------------------------------------------------------------------------
__global__ __launch_bounds__(128) void wmma_gemm_bias_relu(
    const unsigned short* __restrict__ X, int ldx,
    const unsigned short* __restrict__ Wt,     // (E, N, K), expert stride N*K
    const float* __restrict__ bias,            // expert stride N
    unsigned short* __restrict__ Y16,          // bf16 output (or null)
    float* __restrict__ Y32,                   // fp32 output (or null)
    int ldy, int K, int N) {
  const int lane = threadIdx.x & 31;
  const int wave = threadIdx.x >> 5;
  const int e    = blockIdx.z;
  const int row0 = (blockIdx.x * 4 + wave) * 32;   // two 16-row tiles
  const int n0   = blockIdx.y * 64;
  const int lm   = lane & 15;          // M (A) or N (B/C) within tile
  const int lk   = (lane >> 4) * 8;    // K-half selector

  const unsigned short* xrow0 = X + (size_t)(row0 + lm)      * (size_t)ldx;
  const unsigned short* xrow1 = X + (size_t)(row0 + 16 + lm) * (size_t)ldx;
  const unsigned short* wbase =
      Wt + (size_t)e * (size_t)N * (size_t)K + (size_t)(n0 + lm) * (size_t)K;

  v8f acc[2][4] = {};

  for (int k0 = 0; k0 < K; k0 += 32) {
    if (k0 + 32 < K) {
      __builtin_prefetch(xrow0 + k0 + 32, 0, 0);
      __builtin_prefetch(xrow1 + k0 + 32, 0, 0);
    }

    BF16Frag a0, a1;
    a0.q[0] = *(const v4u*)(xrow0 + k0 + lk);        // K = k0+lk   .. +7
    a0.q[1] = *(const v4u*)(xrow0 + k0 + 16 + lk);   // K = k0+16+lk.. +7
    a1.q[0] = *(const v4u*)(xrow1 + k0 + lk);
    a1.q[1] = *(const v4u*)(xrow1 + k0 + 16 + lk);

#pragma unroll
    for (int s = 0; s < 4; ++s) {
      const unsigned short* wr = wbase + (size_t)(s * 16) * (size_t)K;
      BF16Frag b;
      b.q[0] = *(const v4u*)(wr + k0 + lk);
      b.q[1] = *(const v4u*)(wr + k0 + 16 + lk);
      acc[0][s] = __builtin_amdgcn_wmma_f32_16x16x32_bf16(
          false, a0.v, false, b.v, (short)0, acc[0][s], false, false);
      acc[1][s] = __builtin_amdgcn_wmma_f32_16x16x32_bf16(
          false, a1.v, false, b.v, (short)0, acc[1][s], false, false);
    }
  }

  // Epilogue: C layout -> lane 0..15: M = v; lane 16..31: M = v + 8; N = lm.
  const float* bp   = bias + (size_t)e * (size_t)N;
  const int   rhalf = (lane >> 4) * 8;
#pragma unroll
  for (int mt = 0; mt < 2; ++mt) {
#pragma unroll
    for (int s = 0; s < 4; ++s) {
      const int   col = n0 + s * 16 + lm;
      const float bb  = bp[col];
#pragma unroll
      for (int v = 0; v < 8; ++v) {
        float val = acc[mt][s][v] + bb;
        val = val > 0.f ? val : 0.f;
        const size_t yi = (size_t)(row0 + mt * 16 + rhalf + v) * (size_t)ldy +
                          (size_t)e * (size_t)N + col;
        if (Y32) Y32[yi] = val;
        else     Y16[yi] = f2bf(val);
      }
    }
  }
}

// ---------------------------------------------------------------------------
// Gate: softmax(X @ GW + GB) per row. One wave per row, lanes split D,
// __shfl_xor tree reduction, softmax on lane 0. E <= 12.
// ---------------------------------------------------------------------------
__global__ __launch_bounds__(256) void gate_softmax(
    const unsigned short* __restrict__ X, int ldx,
    const float* __restrict__ GW,   // (D, E) row-major
    const float* __restrict__ GB,   // (E)
    float* __restrict__ outw,       // (rows, E) softmaxed gate weights
    int D, int E, int rows) {
  const int row  = (int)((blockIdx.x * blockDim.x + threadIdx.x) >> 5);
  const int lane = threadIdx.x & 31;
  if (row >= rows) return;

  float acc[12];
  for (int e = 0; e < E; ++e) acc[e] = 0.f;

  const unsigned short* xr = X + (size_t)row * (size_t)ldx;
  for (int d = lane; d < D; d += 32) {
    const float  xv = bf2f(xr[d]);
    const float* g  = GW + (size_t)d * (size_t)E;
    for (int e = 0; e < E; ++e) acc[e] += xv * g[e];
  }
  for (int e = 0; e < E; ++e)
    for (int off = 16; off > 0; off >>= 1) acc[e] += __shfl_xor(acc[e], off);

  if (lane == 0) {
    float mx = -3.4e38f;
    for (int e = 0; e < E; ++e) { acc[e] += GB[e]; mx = fmaxf(mx, acc[e]); }
    float sum = 0.f;
    for (int e = 0; e < E; ++e) { acc[e] = __expf(acc[e] - mx); sum += acc[e]; }
    const float inv = 1.f / sum;
    float* o = outw + (size_t)row * (size_t)E;
    for (int e = 0; e < E; ++e) o[e] = acc[e] * inv;
  }
}

// ---------------------------------------------------------------------------
// Weighted combine over expert outputs: out[b,u] = sum_e g[b,e]*E[b,map(e),u].
// Two (offset,count) segments encode concat([task_experts, shared_experts]).
// ---------------------------------------------------------------------------
__global__ __launch_bounds__(256) void gate_combine(
    const float* __restrict__ Wg, int E,
    const unsigned short* __restrict__ Ebuf, int Etot, int U,
    int off0, int cnt0, int off1, int cnt1,
    unsigned short* __restrict__ out16, int total) {
  const int idx = blockIdx.x * blockDim.x + threadIdx.x;
  if (idx >= total) return;
  const int b = idx / U;
  const int u = idx - b * U;
  const float*          g  = Wg + (size_t)b * (size_t)E;
  const unsigned short* eb = Ebuf + (size_t)b * (size_t)Etot * (size_t)U + u;
  float s = 0.f;
  for (int j = 0; j < cnt0; ++j) s += g[j]        * bf2f(eb[(size_t)(off0 + j) * U]);
  for (int j = 0; j < cnt1; ++j) s += g[cnt0 + j] * bf2f(eb[(size_t)(off1 + j) * U]);
  out16[idx] = f2bf(s);
}

// ---------------------------------------------------------------------------
// Host orchestration
// ---------------------------------------------------------------------------
extern "C" void kernel_launch(void* const* d_in, const int* in_sizes, int n_in,
                              void* d_out, int out_size, void* d_ws, size_t ws_size,
                              hipStream_t stream) {
  (void)in_sizes; (void)n_in; (void)out_size; (void)ws_size;
  const int B = PLE_B;

  char*  ws  = (char*)d_ws;
  size_t off = 0;
  auto alloc = [&](size_t bytes) -> char* {
    off = (off + 255) & ~(size_t)255;
    char* p = ws + off;
    off += bytes;
    return p;
  };
  auto a16 = [&](size_t elems) { return (unsigned short*)alloc(elems * 2); };
  auto a32 = [&](size_t elems) { return (float*)alloc(elems * 4); };

  // --- workspace layout ---
  unsigned short* x16    = a16((size_t)B * 1024);
  unsigned short* w_t00  = a16((size_t)4 * 1024 * 512);   // transposed (E,N,K)
  unsigned short* w_t01  = a16((size_t)4 * 1024 * 512);
  unsigned short* w_s0   = a16((size_t)4 * 1024 * 512);
  unsigned short* w_t10  = a16((size_t)4 * 512 * 256);
  unsigned short* w_t11  = a16((size_t)4 * 512 * 256);
  unsigned short* w_s1   = a16((size_t)4 * 512 * 256);
  unsigned short* w_tw00 = a16((size_t)256 * 128);
  unsigned short* w_tw01 = a16((size_t)128 * 64);
  unsigned short* w_tw10 = a16((size_t)256 * 128);
  unsigned short* w_tw11 = a16((size_t)128 * 64);
  unsigned short* e0     = a16((size_t)B * 12 * 512);     // layer-0 experts
  unsigned short* e1     = e0;                            // reuse (consumed)
  unsigned short* task0  = a16((size_t)B * 512);
  unsigned short* task1  = a16((size_t)B * 512);
  unsigned short* shn    = a16((size_t)B * 512);
  float*          gA     = a32((size_t)B * 12);
  float*          gB     = a32((size_t)B * 12);
  float*          gC     = a32((size_t)B * 12);
  unsigned short* o0     = a16((size_t)B * 256);
  unsigned short* o1     = a16((size_t)B * 256);
  unsigned short* h0     = a16((size_t)B * 128);
  unsigned short* h1     = a16((size_t)B * 128);

  auto cvt = [&](int idx, unsigned short* dst, size_t n) {
    cvt_f32_to_bf16<<<dim3((unsigned)((n + 255) / 256)), dim3(256), 0, stream>>>(
        (const float*)d_in[idx], dst, (int)n);
  };
  auto cvtT = [&](int idx, unsigned short* dst, int E, int K, int N) {
    const int total = E * K * N;
    cvt_transpose_f32_bf16<<<dim3((total + 255) / 256), dim3(256), 0, stream>>>(
        (const float*)d_in[idx], dst, K, N, total);
  };
  auto gemm = [&](const unsigned short* X, int ldx, const unsigned short* Wt,
                  const float* bias, unsigned short* Y16, float* Y32, int ldy,
                  int K, int N, int E) {
    dim3 grid(B / 128, N / 64, E);
    wmma_gemm_bias_relu<<<grid, dim3(128), 0, stream>>>(X, ldx, Wt, bias, Y16,
                                                        Y32, ldy, K, N);
  };
  auto gate = [&](const unsigned short* X, int ldx, int gwi, int gbi, float* out,
                  int D, int E) {
    gate_softmax<<<dim3((B * 32) / 256), dim3(256), 0, stream>>>(
        X, ldx, (const float*)d_in[gwi], (const float*)d_in[gbi], out, D, E, B);
  };
  auto combine = [&](const float* Wg, int E, const unsigned short* Ebuf, int U,
                     int off0, int cnt0, int off1, int cnt1, unsigned short* out) {
    const int total = B * U;
    gate_combine<<<dim3((total + 255) / 256), dim3(256), 0, stream>>>(
        Wg, E, Ebuf, 12, U, off0, cnt0, off1, cnt1, out, total);
  };

  // --- conversions (weights transposed to (E,N,K)) ---
  cvt (0,  x16,    (size_t)B * 1024);
  cvtT(1,  w_t00,  4, 1024, 512);
  cvtT(3,  w_t01,  4, 1024, 512);
  cvtT(5,  w_s0,   4, 1024, 512);
  cvtT(13, w_t10,  4, 512, 256);
  cvtT(15, w_t11,  4, 512, 256);
  cvtT(17, w_s1,   4, 512, 256);
  cvtT(23, w_tw00, 1, 256, 128);
  cvtT(25, w_tw01, 1, 128, 64);
  cvtT(27, w_tw10, 1, 256, 128);
  cvtT(29, w_tw11, 1, 128, 64);

  // --- CGC layer 0: experts (E-buffer layout: [t0(0..3), t1(4..7), sh(8..11)]) ---
  gemm(x16, 1024, w_t00, (const float*)d_in[2], e0 + 0 * 512, nullptr, 12 * 512, 1024, 512, 4);
  gemm(x16, 1024, w_t01, (const float*)d_in[4], e0 + 4 * 512, nullptr, 12 * 512, 1024, 512, 4);
  gemm(x16, 1024, w_s0,  (const float*)d_in[6], e0 + 8 * 512, nullptr, 12 * 512, 1024, 512, 4);

  // --- layer 0 gates + combines ---
  gate(x16, 1024, 7,  8,  gA, 1024, 8);    // g0_0 over [t0, sh]
  gate(x16, 1024, 9,  10, gB, 1024, 8);    // g0_1 over [t1, sh]
  gate(x16, 1024, 11, 12, gC, 1024, 12);   // g0_s over [t0, t1, sh]
  combine(gA, 8,  e0, 512, 0, 4, 8, 4, task0);
  combine(gB, 8,  e0, 512, 4, 4, 8, 4, task1);
  combine(gC, 12, e0, 512, 0, 8, 8, 4, shn);

  // --- CGC layer 1: experts ---
  gemm(task0, 512, w_t10, (const float*)d_in[14], e1 + 0 * 256, nullptr, 12 * 256, 512, 256, 4);
  gemm(task1, 512, w_t11, (const float*)d_in[16], e1 + 4 * 256, nullptr, 12 * 256, 512, 256, 4);
  gemm(shn,   512, w_s1,  (const float*)d_in[18], e1 + 8 * 256, nullptr, 12 * 256, 512, 256, 4);

  // --- layer 1 gates + combines (no shared gate in final CGC layer) ---
  gate(task0, 512, 19, 20, gA, 512, 8);
  gate(task1, 512, 21, 22, gB, 512, 8);
  combine(gA, 8, e1, 256, 0, 4, 8, 4, o0);
  combine(gB, 8, e1, 256, 4, 4, 8, 4, o1);

  // --- task towers (final layer writes fp32 straight to d_out) ---
  float* out0 = (float*)d_out;
  float* out1 = (float*)d_out + (size_t)B * 64;
  gemm(o0, 256, w_tw00, (const float*)d_in[24], h0, nullptr, 128, 256, 128, 1);
  gemm(h0, 128, w_tw01, (const float*)d_in[26], nullptr, out0, 64, 128, 64, 1);
  gemm(o1, 256, w_tw10, (const float*)d_in[28], h1, nullptr, 128, 256, 128, 1);
  gemm(h1, 128, w_tw11, (const float*)d_in[30], nullptr, out1, 64, 128, 64, 1);
}
